// FractalModel_14259291422729
// MI455X (gfx1250) — matrix operand order repaired and checked by reading, not compiled, not run on hardware
//
#include <hip/hip_runtime.h>

typedef __attribute__((ext_vector_type(16))) __bf16 v16bf;
typedef __attribute__((ext_vector_type(8)))  float  v8f;
typedef __attribute__((ext_vector_type(8)))  __bf16 bf16x8;
typedef __attribute__((ext_vector_type(4)))  float  f32x4;
typedef __attribute__((ext_vector_type(4)))  __bf16 bf16x4;

// ---------------------------------------------------------------------------
// CDNA5 async global->LDS copy (ASYNCcnt-tracked DMA, bypasses VGPRs).
// LDS operand is the low 32 bits of the generic address (aperture layout puts
// the wave-relative LDS byte offset in addr[31:0]).
// ---------------------------------------------------------------------------
__device__ __forceinline__ void async_copy_b128(const void* gsrc, void* lds) {
    unsigned l = (unsigned)(uintptr_t)lds;
    unsigned long long g = (unsigned long long)(uintptr_t)gsrc;
    asm volatile("global_load_async_to_lds_b128 %0, %1, off"
                 :: "v"(l), "v"(g) : "memory");
}
__device__ __forceinline__ void async_wait0() {
    asm volatile("s_wait_asynccnt 0" ::: "memory");
}

// ---------------------------------------------------------------------------
// fp32 -> bf16 conversion (vectorized x4); n4 = n/4
// ---------------------------------------------------------------------------
__global__ __launch_bounds__(256) void cvt_kernel(const float* __restrict__ src,
                                                  __bf16* __restrict__ dst, int n4) {
    int i = blockIdx.x * 256 + threadIdx.x;
    if (i < n4) {
        f32x4 v = ((const f32x4*)src)[i];
        bf16x4 o;
        for (int e = 0; e < 4; ++e) o[e] = (__bf16)v[e];
        ((bf16x4*)dst)[i] = o;
    }
}

// ---------------------------------------------------------------------------
// Embedding gather: X[m, :] = E[tokens[m], :]   (D = 1024, 2048 blocks)
// ---------------------------------------------------------------------------
__global__ __launch_bounds__(256) void embed_kernel(const int* __restrict__ tokens,
                                                    const float* __restrict__ E,
                                                    float* __restrict__ X) {
    int m = blockIdx.x;
    int tid = threadIdx.x;
    int tok = tokens[m];
    f32x4 v = ((const f32x4*)(E + (size_t)tok * 1024))[tid];
    ((f32x4*)(X + (size_t)m * 1024))[tid] = v;
}

// ---------------------------------------------------------------------------
// RMSNorm (+ optional FiLM gamma/beta), fp32 in -> bf16 out.  D = 1024.
// One block (8 waves) per token row; 4 contiguous elements per thread.
// ---------------------------------------------------------------------------
__global__ __launch_bounds__(256) void rmsnorm_film_kernel(
    const float* __restrict__ X, const float* __restrict__ Wn,
    const float* Gamma, const float* Beta, __bf16* __restrict__ Out) {
    constexpr int D = 1024;
    __shared__ float red[8];
    int m = blockIdx.x;
    int tid = threadIdx.x;
    const float* xr = X + (size_t)m * D;
    f32x4 xv = ((const f32x4*)xr)[tid];
    float ss = xv[0] * xv[0] + xv[1] * xv[1] + xv[2] * xv[2] + xv[3] * xv[3];
    for (int msk = 1; msk < 32; msk <<= 1) ss += __shfl_xor(ss, msk, 32);
    if ((tid & 31) == 0) red[tid >> 5] = ss;
    __syncthreads();
    float tot = 0.f;
    for (int w = 0; w < 8; ++w) tot += red[w];
    float rs = rsqrtf(tot / (float)D + 1.1920929e-07f);
    f32x4 wv = ((const f32x4*)Wn)[tid];
    bf16x4 o;
    if (Gamma) {
        f32x4 gv = ((const f32x4*)Gamma)[tid];
        f32x4 bv = ((const f32x4*)Beta)[tid];
        for (int e = 0; e < 4; ++e)
            o[e] = (__bf16)(xv[e] * rs * wv[e] * gv[e] + bv[e]);
    } else {
        for (int e = 0; e < 4; ++e)
            o[e] = (__bf16)(xv[e] * rs * wv[e]);
    }
    ((bf16x4*)(Out + (size_t)m * D))[tid] = o;
}

// ---------------------------------------------------------------------------
// SwiGLU over fused gate|up buffer: GU = [2048][4096] f32 (g | u halves),
// Tb[m][n] = silu(g)*u, bf16 [2048][2048].  Grid: 4096 blocks x 256.
// ---------------------------------------------------------------------------
__global__ __launch_bounds__(256) void swiglu_kernel(const float* __restrict__ GU,
                                                     __bf16* __restrict__ Tb) {
    int i = blockIdx.x * 256 + threadIdx.x;   // over 2048 * 512 f32x4 chunks
    int m = i >> 9;
    int n4 = i & 511;
    f32x4 g = ((const f32x4*)(GU + (size_t)m * 4096))[n4];
    f32x4 u = ((const f32x4*)(GU + (size_t)m * 4096 + 2048))[n4];
    bf16x4 o;
    for (int e = 0; e < 4; ++e) {
        float sg = g[e] / (1.f + __expf(-g[e]));
        o[e] = (__bf16)(sg * u[e]);
    }
    ((bf16x4*)(Tb + (size_t)m * 2048))[n4] = o;
}

// ---------------------------------------------------------------------------
// WMMA bf16 GEMM: C[M,N] = A[M,K] @ W[N,K]^T
// Block: 256 threads (8 waves), tile 128(M) x 64(N), K-step 32.
// Double-buffered LDS staging via global_load_async_to_lds_b128: tile t+1 DMA
// overlaps WMMA compute of tile t.  Pads of 48 keep rows 16B-aligned.
// EPI: 0 = store f32, 1 = store bf16, 2 = R0 + acc, 3 = R0 + s*(R1-R0+acc)
// ---------------------------------------------------------------------------
template <int EPI>
__global__ __launch_bounds__(256) void gemm_kernel(
    const __bf16* __restrict__ A, const __bf16* __restrict__ W, void* Cout,
    const float* R0, const float* R1, const float* Sc,
    int Mdim, int Ndim, int Kdim) {
    __shared__ __bf16 As[2][128][48];   // row-major A tiles (ping-pong)
    __shared__ __bf16 Bs[2][64][48];    // Bs[n][k] = W[n0+n][k0+k]

    const int tid  = threadIdx.x;
    const int lane = tid & 31;
    const int wave = tid >> 5;
    const int half = lane >> 4;
    const int lcol = lane & 15;
    const int m0 = blockIdx.y * 128;
    const int n0 = blockIdx.x * 64;
    const int wr = wave & 3, wc = wave >> 2;

    const int arow = tid >> 2, ak8 = (tid & 3) * 8;   // staging coords (16B chunks)

    auto stage = [&](int kt, int bufi) {
        int k0 = kt * 32;
        async_copy_b128(A + (size_t)(m0 + arow) * Kdim + k0 + ak8, &As[bufi][arow][ak8]);
        async_copy_b128(A + (size_t)(m0 + arow + 64) * Kdim + k0 + ak8,
                        &As[bufi][arow + 64][ak8]);
        async_copy_b128(W + (size_t)(n0 + arow) * Kdim + k0 + ak8, &Bs[bufi][arow][ak8]);
    };

    v8f acc[2][2] = {};

    stage(0, 0);
    async_wait0();
    __syncthreads();

    const int nk = Kdim >> 5;
    for (int kt = 0; kt < nk; ++kt) {
        const int cur = kt & 1;
        if (kt + 1 < nk) stage(kt + 1, cur ^ 1);   // overlap DMA with compute

        v16bf afr[2], bfr[2];
        for (int mi = 0; mi < 2; ++mi) {
            int row = wr * 32 + mi * 16 + lcol;
            for (int jj = 0; jj < 8; ++jj) {
                int base = ((jj < 4) ? 2 * jj : 2 * jj + 8) + 8 * half;
                afr[mi][2 * jj]     = As[cur][row][base];
                afr[mi][2 * jj + 1] = As[cur][row][base + 1];
            }
        }
        for (int ni = 0; ni < 2; ++ni) {
            int coln = wc * 32 + ni * 16 + lcol;
            for (int jj = 0; jj < 8; ++jj) {
                int kk = ((jj < 4) ? 2 * jj : 2 * jj + 8) + 8 * half;
                bfr[ni][2 * jj]     = Bs[cur][coln][kk];
                bfr[ni][2 * jj + 1] = Bs[cur][coln][kk + 1];
            }
        }
        for (int mi = 0; mi < 2; ++mi)
            for (int ni = 0; ni < 2; ++ni)
                acc[mi][ni] = __builtin_amdgcn_wmma_f32_16x16x32_bf16(
                    false, afr[mi], false, bfr[ni], (short)0, acc[mi][ni], false, false);

        async_wait0();
        __syncthreads();
    }

    // epilogue (C layout: vgpr i -> row i + 8*half, col = lane%16)
    for (int mi = 0; mi < 2; ++mi)
        for (int ni = 0; ni < 2; ++ni)
            for (int i = 0; i < 8; ++i) {
                int row = m0 + wr * 32 + mi * 16 + i + 8 * half;
                int col = n0 + wc * 32 + ni * 16 + lcol;
                size_t idx = (size_t)row * Ndim + col;
                float v = acc[mi][ni][i];
                if constexpr (EPI == 0) {
                    ((float*)Cout)[idx] = v;
                } else if constexpr (EPI == 1) {
                    ((__bf16*)Cout)[idx] = (__bf16)v;
                } else if constexpr (EPI == 2) {
                    ((float*)Cout)[idx] = R0[idx] + v;
                } else {
                    float s = Sc[0];
                    float r0 = R0[idx];
                    ((float*)Cout)[idx] = r0 + s * (R1[idx] - r0 + v);
                }
            }
}

// ---------------------------------------------------------------------------
// Flash attention, causal, T=1024, H=16, hd=64, bf16 in/out, fp32 softmax.
// Grid: (B*H, T/128). Block: 256 threads (8 waves), wave w owns 16 query rows.
// Double-buffered K/V tiles; K staged by async DMA.
// ---------------------------------------------------------------------------
__global__ __launch_bounds__(256) void attn_kernel(const __bf16* __restrict__ QKV,
                                                   __bf16* __restrict__ Oout) {
    constexpr int T = 1024, HD = 64, LDQ = 3072;
    __shared__ __bf16 Ks[2][32][72];      // [token][d]  (pairs contiguous)
    __shared__ __bf16 Vs[2][64][36];      // [d][token]  (token pairs contiguous)
    __shared__ __bf16 Plds[8][16][36];    // per-wave P staging (C -> A layout)

    const int b = blockIdx.x >> 4, head = blockIdx.x & 15;
    const int qtile = blockIdx.y;
    const int tid = threadIdx.x, lane = tid & 31, wave = tid >> 5;
    const int half = lane >> 4, lcol = lane & 15;
    const int qbase = qtile * 128 + wave * 16;     // in-batch position
    const size_t rowbase = (size_t)b * T;

    // Q fragments (A layout), pre-scaled by 1/sqrt(hd)
    v16bf qf[2];
    {
        const __bf16* qp = QKV + (rowbase + qbase + lcol) * LDQ + head * HD;
        for (int f = 0; f < 2; ++f)
            for (int jj = 0; jj < 8; ++jj) {
                int base = ((jj < 4) ? 2 * jj : 2 * jj + 8) + 8 * half + 32 * f;
                qf[f][2 * jj]     = (__bf16)(0.125f * (float)qp[base]);
                qf[f][2 * jj + 1] = (__bf16)(0.125f * (float)qp[base + 1]);
            }
    }

    float mrow[8], lrow[8];
    v8f oacc[4] = {};
    for (int i = 0; i < 8; ++i) { mrow[i] = -1e30f; lrow[i] = 0.f; }

    const int mydiag = qbase >> 5;
    const int jmax = (qtile * 128 + 112) >> 5;

    const int c  = tid >> 3;               // staging token-in-tile 0..31
    const int d8 = (tid & 7) * 8;

    auto stageKV = [&](int j, int bufi) {
        async_copy_b128(QKV + (rowbase + j * 32 + c) * LDQ + 1024 + head * HD + d8,
                        &Ks[bufi][c][d8]);
        const __bf16* vp = QKV + (rowbase + j * 32 + c) * LDQ + 2048 + head * HD + d8;
        bf16x8 vv = *(const bf16x8*)vp;
        for (int e = 0; e < 8; ++e) Vs[bufi][d8 + e][c] = vv[e];
    };

    stageKV(0, 0);
    async_wait0();
    __syncthreads();

    for (int j = 0; j <= jmax; ++j) {
        const int cur = j & 1;
        if (j + 1 <= jmax) stageKV(j + 1, cur ^ 1);   // overlap with compute

        if (j <= mydiag) {                  // wave-uniform branch: EXEC all ones
            v8f s0 = {}, s1 = {};
            for (int ci = 0; ci < 2; ++ci) {
                v8f sc = {};
                for (int f = 0; f < 2; ++f) {
                    v16bf kb;
                    int cc = ci * 16 + lcol;
                    for (int jj = 0; jj < 8; ++jj) {
                        int dd = ((jj < 4) ? 2 * jj : 2 * jj + 8) + 8 * half + 32 * f;
                        kb[2 * jj]     = Ks[cur][cc][dd];
                        kb[2 * jj + 1] = Ks[cur][cc][dd + 1];
                    }
                    sc = __builtin_amdgcn_wmma_f32_16x16x32_bf16(
                        false, qf[f], false, kb, (short)0, sc, false, false);
                }
                if (ci == 0) s0 = sc; else s1 = sc;
            }
            if (j == mydiag) {
                int off = qbase - j * 32;   // 0 or 16
                for (int i = 0; i < 8; ++i) {
                    int rr = i + 8 * half;
                    if (lcol > rr + off)      s0[i] = -1e30f;
                    if (16 + lcol > rr + off) s1[i] = -1e30f;
                }
            }
            float corr[8], p0[8], p1[8];
            for (int i = 0; i < 8; ++i) {
                float t = fmaxf(s0[i], s1[i]);
                for (int msk = 1; msk < 16; msk <<= 1)
                    t = fmaxf(t, __shfl_xor(t, msk, 32));
                float mnew = fmaxf(mrow[i], t);
                p0[i] = __expf(s0[i] - mnew);
                p1[i] = __expf(s1[i] - mnew);
                float rs = p0[i] + p1[i];
                for (int msk = 1; msk < 16; msk <<= 1)
                    rs += __shfl_xor(rs, msk, 32);
                corr[i] = __expf(mrow[i] - mnew);
                lrow[i] = lrow[i] * corr[i] + rs;
                mrow[i] = mnew;
            }
            for (int ni = 0; ni < 4; ++ni)
                for (int i = 0; i < 8; ++i) oacc[ni][i] *= corr[i];

            for (int i = 0; i < 8; ++i) {
                int rr = i + 8 * half;
                Plds[wave][rr][lcol]      = (__bf16)p0[i];
                Plds[wave][rr][16 + lcol] = (__bf16)p1[i];
            }
            v16bf pf;
            for (int jj = 0; jj < 8; ++jj) {
                int base = ((jj < 4) ? 2 * jj : 2 * jj + 8) + 8 * half;
                pf[2 * jj]     = Plds[wave][lcol][base];
                pf[2 * jj + 1] = Plds[wave][lcol][base + 1];
            }
            for (int ni = 0; ni < 4; ++ni) {
                v16bf vb;
                int dn = ni * 16 + lcol;
                for (int jj = 0; jj < 8; ++jj) {
                    int kk = ((jj < 4) ? 2 * jj : 2 * jj + 8) + 8 * half;
                    vb[2 * jj]     = Vs[cur][dn][kk];
                    vb[2 * jj + 1] = Vs[cur][dn][kk + 1];
                }
                oacc[ni] = __builtin_amdgcn_wmma_f32_16x16x32_bf16(
                    false, pf, false, vb, (short)0, oacc[ni], false, false);
            }
        }
        async_wait0();
        __syncthreads();
    }

    for (int ni = 0; ni < 4; ++ni)
        for (int i = 0; i < 8; ++i) {
            int tok = qbase + i + 8 * half;
            int dn = head * HD + ni * 16 + lcol;
            Oout[(rowbase + tok) * 1024 + dn] = (__bf16)(oacc[ni][i] / lrow[i]);
        }
}

// ---------------------------------------------------------------------------
// Host orchestration
// ---------------------------------------------------------------------------
extern "C" void kernel_launch(void* const* d_in, const int* in_sizes, int n_in,
                              void* d_out, int out_size, void* d_ws, size_t ws_size,
                              hipStream_t stream) {
    const int*   tokens  = (const int*)d_in[0];
    const float* embed_w = (const float*)d_in[1];
    const float* qkv_w   = (const float*)d_in[2];
    const float* o_w     = (const float*)d_in[3];
    const float* gate_w  = (const float*)d_in[4];
    const float* up_w    = (const float*)d_in[5];
    const float* down_w  = (const float*)d_in[6];
    const float* norm1_w = (const float*)d_in[7];
    const float* norm2_w = (const float*)d_in[8];
    const float* sgamma  = (const float*)d_in[9];
    const float* sbeta   = (const float*)d_in[10];
    const float* iters   = (const float*)d_in[11];
    const float* norm_w  = (const float*)d_in[12];
    const float* lm_w    = (const float*)d_in[13];
    float* outp = (float*)d_out;

    char* ws = (char*)d_ws;
    const size_t MB = 1024ull * 1024ull;
    float*  x     = (float*)(ws + 0 * MB);    // [2048,1024] f32     8 MB
    float*  x1    = (float*)(ws + 8 * MB);    // [2048,1024] f32     8 MB
    __bf16* hbuf  = (__bf16*)(ws + 16 * MB);  // [2048,1024] bf16    4 MB
    __bf16* qkvb  = (__bf16*)(ws + 20 * MB);  // [2048,3072] bf16   12 MB
    __bf16* attnb = (__bf16*)(ws + 32 * MB);  // [2048,1024] bf16    4 MB
    float*  gbuf  = (float*)(ws + 36 * MB);   // [2048,4096] f32    32 MB (g|u)
    __bf16* tbuf  = (__bf16*)(ws + 68 * MB);  // [2048,2048] bf16    8 MB
    __bf16* wqkv  = (__bf16*)(ws + 76 * MB);  //  6 MB
    __bf16* wo    = (__bf16*)(ws + 82 * MB);  //  2 MB
    __bf16* wgu   = (__bf16*)(ws + 84 * MB);  //  8 MB (gate ++ up, [4096,1024])
    __bf16* wd    = (__bf16*)(ws + 92 * MB);  //  4 MB
    __bf16* wlm   = (__bf16*)(ws + 96 * MB);  // 64 MB

    auto cvt = [&](const float* s, __bf16* d, int n) {
        cvt_kernel<<<(n / 4 + 255) / 256, 256, 0, stream>>>(s, d, n / 4);
    };
    cvt(qkv_w, wqkv, 3 * 1024 * 1024);
    cvt(o_w,   wo,   1024 * 1024);
    cvt(gate_w, wgu, 2048 * 1024);                     // gate -> first 2048 rows
    cvt(up_w,   wgu + 2048ull * 1024ull, 2048 * 1024); // up   -> last 2048 rows
    cvt(down_w, wd,  1024 * 2048);
    cvt(lm_w,   wlm, 32000 * 1024);

    embed_kernel<<<2048, 256, 0, stream>>>(tokens, embed_w, x);

    for (int ell = 0; ell < 32; ++ell) {
        const float* gamma = sgamma + (ell / 8) * 1024;
        const float* beta  = sbeta + (ell / 8) * 1024;
        const float* sptr  = iters + ell;

        rmsnorm_film_kernel<<<2048, 256, 0, stream>>>(x, norm1_w, gamma, beta, hbuf);
        gemm_kernel<1><<<dim3(48, 16), 256, 0, stream>>>(hbuf, wqkv, qkvb,
            nullptr, nullptr, nullptr, 2048, 3072, 1024);
        attn_kernel<<<dim3(32, 8), 256, 0, stream>>>(qkvb, attnb);
        gemm_kernel<2><<<dim3(16, 16), 256, 0, stream>>>(attnb, wo, x1,
            x, nullptr, nullptr, 2048, 1024, 1024);
        rmsnorm_film_kernel<<<2048, 256, 0, stream>>>(x1, norm2_w, gamma, beta, hbuf);
        gemm_kernel<0><<<dim3(64, 16), 256, 0, stream>>>(hbuf, wgu, gbuf,
            nullptr, nullptr, nullptr, 2048, 4096, 1024);
        swiglu_kernel<<<4096, 256, 0, stream>>>(gbuf, tbuf);
        gemm_kernel<3><<<dim3(16, 16), 256, 0, stream>>>(tbuf, wd, x,
            x, x1, sptr, 2048, 1024, 2048);
    }

    rmsnorm_film_kernel<<<2048, 256, 0, stream>>>(x, norm_w, nullptr, nullptr, hbuf);
    gemm_kernel<0><<<dim3(500, 16), 256, 0, stream>>>(hbuf, wlm, outp,
        nullptr, nullptr, nullptr, 2048, 32000, 1024);
}
